// QuantLinearTorch_51599737094843
// MI455X (gfx1250) — matrix operand order repaired and checked by reading, not compiled, not run on hardware
//
#include <hip/hip_runtime.h>

// ---- problem constants (match reference) ----
#define IN_F   4096
#define OUT_F  11008
#define BATCH  4
#define GROUP  128
#define KSEG   1024                 // K split for occupancy
#define NSEG   (IN_F / KSEG)        // 4
#define TILE_N 16
#define TILES_PER_WAVE 4
#define COLS_PER_WAVE  (TILE_N * TILES_PER_WAVE)  // 64
#define NCHUNK (IN_F / 32)          // 128 K-chunks of 32

typedef _Float16 h2   __attribute__((ext_vector_type(2)));
typedef _Float16 v16h __attribute__((ext_vector_type(16)));
typedef float    v8f  __attribute__((ext_vector_type(8)));

// -----------------------------------------------------------------------------
// Prep: x[4,4096] f32 -> A-workspace f16, padded to 16 rows, K-permuted so the
// main kernel's per-lane A fragment is a contiguous 32B load that matches the
// nibble-pair order of the B fragments (K is a contraction index: permuting
// K identically in A and B leaves the WMMA result unchanged).
//
// aws[((m*NCHUNK + c)*32) + p] = x_f16[m][c*32 + aperm[p]]   (0 for m>=4)
// aperm[p]: g=p>>3 selects K-base {0,8,16,24}; i=p&7 -> (i>>1) + ((i&1)<<2)
// -----------------------------------------------------------------------------
__global__ void build_a_kernel(const float* __restrict__ x,
                               _Float16* __restrict__ aws) {
    int idx = blockIdx.x * blockDim.x + threadIdx.x;   // 0 .. 16*4096-1
    int m   = idx >> 12;                               // row 0..15
    int rem = idx & 4095;
    int c   = rem >> 5;                                // K-chunk
    int p   = rem & 31;                                // permuted pos in chunk
    int g   = p >> 3;
    int i   = p & 7;
    int base = g << 3;                                 // 0,8,16,24
    int koff = base + (i >> 1) + ((i & 1) << 2);       // 0,4,1,5,2,6,3,7 pattern
    float v = (m < BATCH) ? x[m * IN_F + c * 32 + koff] : 0.0f;
    aws[idx] = (_Float16)v;
}

// -----------------------------------------------------------------------------
// Main: one wave per block. Wave owns 64 output columns (4 WMMA tiles) and a
// 1024-deep K segment. Streams qweight coalesced, dequants nibbles to f16 via
// the 0x6400 exponent trick + v_pk_fma_f16, accumulates with
// v_wmma_f32_16x16x32_f16, writes split-K partials to workspace.
// -----------------------------------------------------------------------------
__global__ void __launch_bounds__(32)
qgemm_wmma_kernel(const int* __restrict__ qw,
                  const float* __restrict__ scales,
                  const float* __restrict__ zeros,
                  const _Float16* __restrict__ aws,
                  float* __restrict__ partial) {
    const int lane    = threadIdx.x;     // 0..31
    const int laneLo  = lane & 15;
    const int hi      = lane >> 4;       // 0/1 half-wave
    const int colBase = blockIdx.x * COLS_PER_WAVE;
    const int seg     = blockIdx.y;      // 0..NSEG-1
    const int kBase   = seg * KSEG;

    v8f acc[TILES_PER_WAVE];
    {
        v8f zeroAcc = {};
        #pragma unroll
        for (int t = 0; t < TILES_PER_WAVE; ++t) acc[t] = zeroAcc;
    }

    h2 s2[TILES_PER_WAVE];   // packed scale
    h2 nz2[TILES_PER_WAVE];  // packed -zero

    // Lane's A base: row m = lane%16, half-offset hi*16 within each 32-half chunk
    const _Float16* aRow = aws + (size_t)laneLo * NCHUNK * 32 + hi * 16;

    const h2 k1024 = {(_Float16)1024.0f, (_Float16)1024.0f};

    #pragma unroll 1
    for (int it = 0; it < KSEG / 32; ++it) {
        const int k0 = kBase + it * 32;
        const int c  = k0 >> 5;

        // reload per-group scale/zero every GROUP(=128) K, i.e. every 4 steps
        if ((k0 & (GROUP - 1)) == 0) {
            const int g = k0 >> 7;
            #pragma unroll
            for (int t = 0; t < TILES_PER_WAVE; ++t) {
                const int col = colBase + t * TILE_N + laneLo;
                _Float16 sh = (_Float16)scales[g * OUT_F + col];
                _Float16 zh = (_Float16)(-zeros[g * OUT_F + col]);
                s2[t]  = (h2){sh, sh};
                nz2[t] = (h2){zh, zh};
            }
        }

        // A fragment: 16 contiguous halfs (32B) — L2/WGP$-hot
        const v16h afrag = *(const v16h*)(aRow + (size_t)c * 32);

        // qweight rows for this K-step: lanes 0-15 -> rows r,r+1 (K 0..15);
        // lanes 16-31 -> rows r+2,r+3 (K 16..31)
        const int r = (k0 >> 3) + 2 * hi;

        // prefetch next K-step's rows (speculative; dropped if OOB)
        __builtin_prefetch(qw + (size_t)(r + 4) * OUT_F + colBase + laneLo, 0, 3);
        __builtin_prefetch(qw + (size_t)(r + 5) * OUT_F + colBase + laneLo, 0, 3);

        #pragma unroll
        for (int t = 0; t < TILES_PER_WAVE; ++t) {
            const int col = colBase + t * TILE_N + laneLo;
            const unsigned w0 = (unsigned)qw[(size_t)r       * OUT_F + col];
            const unsigned w1 = (unsigned)qw[(size_t)(r + 1) * OUT_F + col];

            union { v16h v; h2 p[8]; } b;
            #pragma unroll
            for (int d = 0; d < 2; ++d) {
                const unsigned w = d ? w1 : w0;
                #pragma unroll
                for (int s = 0; s < 4; ++s) {
                    // packed (1024+q_s, 1024+q_{s+4}) as two f16
                    unsigned bits = ((w >> (4 * s)) & 0x000F000Fu) | 0x64006400u;
                    h2 hq = __builtin_bit_cast(h2, bits);
                    h2 q  = hq - k1024;                        // exact
                    // W = q*scale - zero, single-rounded (v_pk_fma_f16)
                    b.p[d * 4 + s] = __builtin_elementwise_fma(q, s2[t], nz2[t]);
                }
            }
            acc[t] = __builtin_amdgcn_wmma_f32_16x16x32_f16(
                         false, afrag, false, b.v,
                         (short)0, acc[t], false, false);
        }
    }

    // C layout: VGPR i, lanes 0-15 -> row M=i, col N=lane. Rows 0..3 are real.
    if (lane < 16) {
        #pragma unroll
        for (int i = 0; i < BATCH; ++i) {
            #pragma unroll
            for (int t = 0; t < TILES_PER_WAVE; ++t) {
                partial[(size_t)(seg * BATCH + i) * OUT_F
                        + colBase + t * TILE_N + laneLo] = acc[t][i];
            }
        }
    }
}

// -----------------------------------------------------------------------------
// Deterministic split-K reduction: out[b][n] = sum_seg partial[seg][b][n]
// -----------------------------------------------------------------------------
__global__ void reduce_kernel(const float* __restrict__ partial,
                              float* __restrict__ out) {
    int idx = blockIdx.x * blockDim.x + threadIdx.x;
    if (idx >= BATCH * OUT_F) return;
    int b = idx / OUT_F;
    int n = idx - b * OUT_F;
    float s = 0.0f;
    #pragma unroll
    for (int sg = 0; sg < NSEG; ++sg)
        s += partial[(size_t)(sg * BATCH + b) * OUT_F + n];
    out[idx] = s;
}

extern "C" void kernel_launch(void* const* d_in, const int* in_sizes, int n_in,
                              void* d_out, int out_size, void* d_ws, size_t ws_size,
                              hipStream_t stream) {
    const float* x      = (const float*)d_in[0];
    const int*   qw     = (const int*)  d_in[1];
    const float* scales = (const float*)d_in[2];
    const float* zeros  = (const float*)d_in[3];
    float*       out    = (float*)d_out;

    // workspace layout: [A f16 16x4096 = 128KB][partials 4x4x11008 f32 = 688KB]
    _Float16* aws     = (_Float16*)d_ws;
    float*    partial = (float*)((char*)d_ws + (size_t)16 * IN_F * sizeof(_Float16));

    build_a_kernel<<<(16 * IN_F) / 256, 256, 0, stream>>>(x, aws);

    dim3 grid(OUT_F / COLS_PER_WAVE, NSEG);   // 172 x 4 waves
    qgemm_wmma_kernel<<<grid, 32, 0, stream>>>(qw, scales, zeros, aws, partial);

    reduce_kernel<<<(BATCH * OUT_F + 255) / 256, 256, 0, stream>>>(partial, out);
}
